// GAT_7095285973830
// MI455X (gfx1250) — compile-verified, matrix operand order checked by reading, not run on hardware
//
#include <hip/hip_runtime.h>

// ---------------------------------------------------------------------------
// GAT (3-layer) for MI455X / gfx1250.
//   - Dense projections: bf16 WMMA (v_wmma_f32_16x16x32_bf16), f32 accumulate,
//     double-buffered async global->LDS staging (ASYNCcnt pipeline)
//   - Edge softmax + aggregation: atomic segment ops (roofline: HBM-bound)
// ---------------------------------------------------------------------------

#define NN      50000
#define NE      800000
#define INF_    128
#define HID     64
#define HEADS   4
#define NCLS    256
#define NEG_SLOPE 0.2f

typedef __attribute__((ext_vector_type(16))) __bf16    bf16x16;
typedef __attribute__((ext_vector_type(8)))  float     floatx8;
typedef __attribute__((ext_vector_type(4)))  unsigned  uintx4;
typedef int v4i_ __attribute__((vector_size(4 * sizeof(int))));   // matches builtin param

union FragBF { bf16x16 v; uintx4 q[2]; };

#ifndef __has_builtin
#define __has_builtin(x) 0
#endif
#if __has_builtin(__builtin_amdgcn_global_load_async_to_lds_b128)
#define HAS_ASYNC_LDS 1
#else
#define HAS_ASYNC_LDS 0
#endif

#define AS_GLOBAL __attribute__((address_space(1)))
#define AS_LOCAL  __attribute__((address_space(3)))

// global(bf16, 16B) -> LDS, async when the toolchain exposes the gfx1250 path
__device__ __forceinline__ void cp_b128_g2l(const unsigned short* g, unsigned short* l) {
#if HAS_ASYNC_LDS
    __builtin_amdgcn_global_load_async_to_lds_b128(
        (AS_GLOBAL v4i_*)(void*)g, (AS_LOCAL v4i_*)l, 0, 0);
#else
    *(uintx4*)l = *(const uintx4*)g;
#endif
}

__device__ __forceinline__ void wait_async_copies() {
#if HAS_ASYNC_LDS
#if __has_builtin(__builtin_amdgcn_s_wait_asynccnt)
    __builtin_amdgcn_s_wait_asynccnt(0);
#else
    asm volatile("s_wait_asynccnt 0x0" ::: "memory");
#endif
#endif
}

__device__ __forceinline__ unsigned short f2bf(float f) {
    unsigned u = __float_as_uint(f);
    u += 0x7FFFu + ((u >> 16) & 1u);          // round-to-nearest-even
    return (unsigned short)(u >> 16);
}
__device__ __forceinline__ unsigned fflip(float f) {   // monotonic float->uint
    unsigned u = __float_as_uint(f);
    return (u >> 31) ? ~u : (u | 0x80000000u);
}
__device__ __forceinline__ float funflip(unsigned u) {
    u = (u >> 31) ? (u & 0x7FFFFFFFu) : ~u;
    return __uint_as_float(u);
}

// ---------------------------------------------------------------------------
// casts
// ---------------------------------------------------------------------------
__global__ void k_cast_bf16(const float* __restrict__ s, unsigned short* __restrict__ d, long n) {
    long i = blockIdx.x * (long)blockDim.x + threadIdx.x;
    if (i < n) d[i] = f2bf(s[i]);
}

// W [K,Nout] f32 -> Wt [Nout,K] bf16 (transposed so GEMM B-tiles are contiguous)
__global__ void k_cast_bf16_t(const float* __restrict__ W, unsigned short* __restrict__ Wt,
                              int K, int Nout) {
    long i = blockIdx.x * (long)blockDim.x + threadIdx.x;   // i = k*Nout + n
    if (i >= (long)K * Nout) return;
    int n = (int)(i % Nout);
    int k = (int)(i / Nout);
    Wt[(size_t)n * K + k] = f2bf(W[i]);
}

__global__ void k_fill_f32(float* __restrict__ p, float v, long n) {
    long i = blockIdx.x * (long)blockDim.x + threadIdx.x;
    if (i < n) p[i] = v;
}

// ---------------------------------------------------------------------------
// Tiled bf16 WMMA GEMM: C[M,Nout] = A[M,K] * Bt[Nout,K]^T   (A,Bt bf16; C f32)
// Block = 128 threads (4 waves). Block tile 64x128, K step 32, double-buffered
// async global->LDS staging. Wave w: rows [16w,16w+16) x 128 cols -> 8 WMMAs
// per K-step off a single A fragment (ISA 7.12.2 16-bit layouts).
// ---------------------------------------------------------------------------
__global__ void __launch_bounds__(128)
k_gemm_bf16(const unsigned short* __restrict__ A,    // [M,K]
            const unsigned short* __restrict__ Bt,   // [Nout,K]
            float* __restrict__ C,                   // [M,Nout]
            int M, int K, int Nout)
{
    __shared__ unsigned short lsA[2][64 * 32];    // [row][k]
    __shared__ unsigned short lsB[2][128 * 32];   // [n][k]

    const int tid   = threadIdx.x;
    const int wave  = tid >> 5;
    const int lane  = tid & 31;
    const int m16   = lane & 15;
    const int hi    = lane >> 4;
    const int khalf = hi * 8;
    const int m0    = blockIdx.x * 64;
    const int n0    = blockIdx.y * 128;

    const floatx8 z8 = {0.f,0.f,0.f,0.f,0.f,0.f,0.f,0.f};
    floatx8 acc[8];
    #pragma unroll
    for (int i = 0; i < 8; ++i) acc[i] = z8;

    // stage one 64x32 A tile + 128x32 B tile into LDS buffer `buf`
    auto stage = [&](int buf, int k0) {
        #pragma unroll
        for (int j = 0; j < 2; ++j) {                 // A: 256 x b128 chunks
            int lin = tid + j * 128;
            int row = lin >> 2;                       // 4 chunks per 32-half row
            int qc  = lin & 3;
            int ar  = m0 + row;
            if (ar >= M) ar = M - 1;                  // pad rows never stored
            cp_b128_g2l(A + (size_t)ar * K + k0 + qc * 8,
                        &lsA[buf][row * 32 + qc * 8]);
        }
        #pragma unroll
        for (int j = 0; j < 4; ++j) {                 // B: 512 x b128 chunks
            int lin = tid + j * 128;
            int n   = lin >> 2;
            int qc  = lin & 3;
            cp_b128_g2l(Bt + (size_t)(n0 + n) * K + k0 + qc * 8,
                        &lsB[buf][n * 32 + qc * 8]);
        }
    };

    stage(0, 0);
    const int nk = K >> 5;
    for (int i = 0; i < nk; ++i) {
        const int cur = i & 1;
        wait_async_copies();      // own wave's staged tile is in LDS
        __syncthreads();          // everyone's staged tile is in LDS / prev reads done
        if (i + 1 < nk)
            stage(cur ^ 1, (i + 1) << 5);   // overlap next stage with compute

        FragBF fa;
        const unsigned short* pa = &lsA[cur][(wave * 16 + m16) * 32];
        fa.q[0] = *(const uintx4*)(pa + khalf);
        fa.q[1] = *(const uintx4*)(pa + 16 + khalf);

        #pragma unroll
        for (int ns = 0; ns < 8; ++ns) {
            FragBF fb;
            const unsigned short* pb = &lsB[cur][(ns * 16 + m16) * 32];
            fb.q[0] = *(const uintx4*)(pb + khalf);
            fb.q[1] = *(const uintx4*)(pb + 16 + khalf);
            acc[ns] = __builtin_amdgcn_wmma_f32_16x16x32_bf16(
                false, fa.v, false, fb.v, (short)0, acc[ns], false, false);
        }
    }

    // epilogue: C/D layout — VGPR i: rows i (lanes 0-15), i+8 (lanes 16-31)
    #pragma unroll
    for (int ns = 0; ns < 8; ++ns) {
        #pragma unroll
        for (int i = 0; i < 8; ++i) {
            int row = m0 + wave * 16 + hi * 8 + i;
            int col = n0 + ns * 16 + m16;
            if (row < M) C[(size_t)row * Nout + col] = acc[ns][i];
        }
    }
}

// ---------------------------------------------------------------------------
// per-(node, head) attention score halves; also init segment max / denom
// ---------------------------------------------------------------------------
__global__ void k_scores(const float* __restrict__ z, const float* __restrict__ al,
                         const float* __restrict__ ar, float* __restrict__ el,
                         float* __restrict__ er, unsigned* __restrict__ emax,
                         float* __restrict__ den, int F)
{
    int idx = blockIdx.x * blockDim.x + threadIdx.x;   // (n, h)
    if (idx >= NN * HEADS) return;
    int h = idx & (HEADS - 1);
    int n = idx >> 2;
    const float* zp  = z + (size_t)n * HEADS * F + (size_t)h * F;
    const float* alp = al + h * F;
    const float* arp = ar + h * F;
    float sl = 0.f, sr = 0.f;
    for (int f = 0; f < F; ++f) { float v = zp[f]; sl += v * alp[f]; sr += v * arp[f]; }
    el[idx] = sl;
    er[idx] = sr;
    emax[idx] = 0u;      // fflip(x) > 0 for all finite x, so 0 == -inf sentinel
    den[idx]  = 0.f;
}

// pass 1: e = leaky_relu(el[src] + er[dst]); segment max via monotone atomicMax
__global__ void k_edge_max(const int* __restrict__ src, const int* __restrict__ dst,
                           const float* __restrict__ el, const float* __restrict__ er,
                           float* __restrict__ eattn, unsigned* __restrict__ emax)
{
    long idx = blockIdx.x * (long)blockDim.x + threadIdx.x;
    if (idx >= (long)NE * HEADS) return;
    int h = (int)(idx & 3);
    int e = (int)(idx >> 2);
    int s = src[e], d = dst[e];
    float x = el[s * HEADS + h] + er[d * HEADS + h];
    x = x > 0.f ? x : NEG_SLOPE * x;
    eattn[idx] = x;
    atomicMax(&emax[d * HEADS + h], fflip(x));
}

// pass 2: ex = exp(e - max); segment sum
__global__ void k_edge_exp(const int* __restrict__ dst, float* __restrict__ eattn,
                           const unsigned* __restrict__ emax, float* __restrict__ den)
{
    long idx = blockIdx.x * (long)blockDim.x + threadIdx.x;
    if (idx >= (long)NE * HEADS) return;
    int h = (int)(idx & 3);
    int e = (int)(idx >> 2);
    int d = dst[e];
    float m  = funflip(emax[d * HEADS + h]);
    float ex = __expf(eattn[idx] - m);
    eattn[idx] = ex;
    atomicAdd(&den[d * HEADS + h], ex);
}

// pass 3: acc[dst] += alpha * z[src]   (float4-vectorized scatter)
__global__ void k_edge_msg(const int* __restrict__ src, const int* __restrict__ dst,
                           const float* __restrict__ eattn, const float* __restrict__ den,
                           const float* __restrict__ z, float* __restrict__ acc, int F)
{
    long idx = blockIdx.x * (long)blockDim.x + threadIdx.x;
    int fq = F >> 2;
    long total = (long)NE * HEADS * fq;
    if (idx >= total) return;
    int  f4 = (int)(idx % fq);
    long t  = idx / fq;
    int  h  = (int)(t & 3);
    int  e  = (int)(t >> 2);
    int  s = src[e], d = dst[e];
    float dn    = den[d * HEADS + h];
    float alpha = eattn[(long)e * HEADS + h] / fmaxf(dn, 1e-9f);
    const float4 v = *((const float4*)(z + (size_t)s * HEADS * F + (size_t)h * F) + f4);
    float* ap = acc + (size_t)d * HEADS * F + (size_t)h * F + f4 * 4;
    atomicAdd(ap + 0, v.x * alpha);
    atomicAdd(ap + 1, v.y * alpha);
    atomicAdd(ap + 2, v.z * alpha);
    atomicAdd(ap + 3, v.w * alpha);
}

// bias + ELU, emit bf16 activations for next layer's GEMM
__global__ void k_bias_elu_bf16(const float* __restrict__ acc, const float* __restrict__ b,
                                unsigned short* __restrict__ hbf, int HF)
{
    long idx = blockIdx.x * (long)blockDim.x + threadIdx.x;
    if (idx >= (long)NN * HF) return;
    int c = (int)(idx % HF);
    float v = acc[idx] + b[c];
    v = v > 0.f ? v : (__expf(v) - 1.f);
    hbf[idx] = f2bf(v);
}

// final: mean over heads of (acc + b2)
__global__ void k_final(const float* __restrict__ acc, const float* __restrict__ b2,
                        float* __restrict__ out)
{
    long idx = blockIdx.x * (long)blockDim.x + threadIdx.x;
    if (idx >= (long)NN * NCLS) return;
    int  c = (int)(idx % NCLS);
    long n = idx / NCLS;
    float s = 0.f;
    #pragma unroll
    for (int h = 0; h < HEADS; ++h)
        s += acc[n * (HEADS * NCLS) + h * NCLS + c] + b2[h * NCLS + c];
    out[idx] = s * 0.25f;
}

// ---------------------------------------------------------------------------
static void run_gat_layer(const unsigned short* hbf_in, int K,
                          const float* W, const float* al, const float* ar, int F,
                          unsigned short* Wbf, float* z, float* acc,
                          float* el, float* er, unsigned* emax, float* den, float* eattn,
                          const int* src, const int* dst, hipStream_t stream)
{
    const int Nout = HEADS * F;
    long nW = (long)K * Nout;
    k_cast_bf16_t<<<(unsigned)((nW + 255) / 256), 256, 0, stream>>>(W, Wbf, K, Nout);

    dim3 g((NN + 63) / 64, Nout / 128);
    k_gemm_bf16<<<g, 128, 0, stream>>>(hbf_in, Wbf, z, NN, K, Nout);

    int nh = NN * HEADS;
    k_scores<<<(nh + 255) / 256, 256, 0, stream>>>(z, al, ar, el, er, emax, den, F);

    long nacc = (long)NN * Nout;
    k_fill_f32<<<(unsigned)((nacc + 255) / 256), 256, 0, stream>>>(acc, 0.f, nacc);

    long neh = (long)NE * HEADS;
    k_edge_max<<<(unsigned)((neh + 255) / 256), 256, 0, stream>>>(src, dst, el, er, eattn, emax);
    k_edge_exp<<<(unsigned)((neh + 255) / 256), 256, 0, stream>>>(dst, eattn, emax, den);

    long nmsg = neh * (F / 4);
    k_edge_msg<<<(unsigned)((nmsg + 255) / 256), 256, 0, stream>>>(src, dst, eattn, den, z, acc, F);
}

extern "C" void kernel_launch(void* const* d_in, const int* in_sizes, int n_in,
                              void* d_out, int out_size, void* d_ws, size_t ws_size,
                              hipStream_t stream)
{
    (void)in_sizes; (void)n_in; (void)out_size; (void)ws_size;

    const float* feat = (const float*)d_in[0];
    const float* W0   = (const float*)d_in[1];
    const float* al0  = (const float*)d_in[2];
    const float* ar0  = (const float*)d_in[3];
    const float* b0   = (const float*)d_in[4];
    const float* W1   = (const float*)d_in[5];
    const float* al1  = (const float*)d_in[6];
    const float* ar1  = (const float*)d_in[7];
    const float* b1   = (const float*)d_in[8];
    const float* W2   = (const float*)d_in[9];
    const float* al2  = (const float*)d_in[10];
    const float* ar2  = (const float*)d_in[11];
    const float* b2   = (const float*)d_in[12];
    const int*   src  = (const int*)d_in[13];
    const int*   dst  = (const int*)d_in[14];
    float*       out  = (float*)d_out;

    char* ws = (char*)d_ws;
    size_t off = 0;
    auto take = [&](size_t bytes) -> char* {
        char* p = ws + off;
        off = (off + bytes + 255) & ~(size_t)255;
        return p;
    };
    unsigned short* hbf   = (unsigned short*)take((size_t)NN * 256 * sizeof(unsigned short));
    unsigned short* Wbf   = (unsigned short*)take((size_t)256 * 1024 * sizeof(unsigned short));
    float*          z     = (float*)take((size_t)NN * 1024 * sizeof(float));
    float*          acc   = (float*)take((size_t)NN * 1024 * sizeof(float));
    float*          el    = (float*)take((size_t)NN * HEADS * sizeof(float));
    float*          er    = (float*)take((size_t)NN * HEADS * sizeof(float));
    unsigned*       emax  = (unsigned*)take((size_t)NN * HEADS * sizeof(unsigned));
    float*          den   = (float*)take((size_t)NN * HEADS * sizeof(float));
    float*          eattn = (float*)take((size_t)NE * HEADS * sizeof(float));

    // layer 0: input feat [NN, 128] -> bf16
    long nf = (long)NN * INF_;
    k_cast_bf16<<<(unsigned)((nf + 255) / 256), 256, 0, stream>>>(feat, hbf, nf);
    run_gat_layer(hbf, INF_, W0, al0, ar0, HID, Wbf, z, acc, el, er, emax, den, eattn, src, dst, stream);
    long nh1 = (long)NN * (HEADS * HID);
    k_bias_elu_bf16<<<(unsigned)((nh1 + 255) / 256), 256, 0, stream>>>(acc, b0, hbf, HEADS * HID);

    // layer 1
    run_gat_layer(hbf, HEADS * HID, W1, al1, ar1, HID, Wbf, z, acc, el, er, emax, den, eattn, src, dst, stream);
    k_bias_elu_bf16<<<(unsigned)((nh1 + 255) / 256), 256, 0, stream>>>(acc, b1, hbf, HEADS * HID);

    // layer 2 (no activation, mean over heads)
    run_gat_layer(hbf, HEADS * HID, W2, al2, ar2, NCLS, Wbf, z, acc, el, er, emax, den, eattn, src, dst, stream);
    long nout = (long)NN * NCLS;
    k_final<<<(unsigned)((nout + 255) / 256), 256, 0, stream>>>(acc, b2, out);
}
